// LSTMModel_23733989278213
// MI455X (gfx1250) — matrix operand order repaired and checked by reading, not compiled, not run on hardware
//
#include <hip/hip_runtime.h>

// LSTM: B=1024, S=1024, H=128. Recurrence is serial over S; parallel over B.
// 64 workgroups x 256 threads (8 wave32). Each WG owns 16 batch rows.
// Each wave owns a 16-wide hidden slice and does 16 v_wmma_f32_16x16x32_f16
// per timestep (4 gates x K=128 in 4 chunks of 32). Wh held in registers as
// WMMA B fragments for all 1024 steps; h double-buffered in LDS; c in VGPRs.

typedef _Float16 v16h __attribute__((ext_vector_type(16)));
typedef _Float16 v8h  __attribute__((ext_vector_type(8)));
typedef float    v8f  __attribute__((ext_vector_type(8)));

#define BQ 1024
#define SQ 1024
#define HQ 128
#define G4H (4 * HQ)
#define CHUNK 128

__device__ __forceinline__ float fast_sigmoid(float x) {
    // 1/(1+exp(-x)); exp overflow -> inf -> rcp -> 0 (correct saturation)
    float e = __expf(-x);
    return __builtin_amdgcn_rcpf(1.0f + e);
}

__device__ __forceinline__ float fast_tanh(float x) {
    // 1 - 2/(exp(2x)+1); saturates correctly at +-1
    float e = __expf(2.0f * x);
    return 1.0f - 2.0f * __builtin_amdgcn_rcpf(e + 1.0f);
}

__global__ __launch_bounds__(256) void lstm_wmma_kernel(
    const float* __restrict__ x,   // [B,S]
    const float* __restrict__ Wi,  // [1,4H]
    const float* __restrict__ Wh,  // [H,4H]
    const float* __restrict__ b,   // [4H]
    const float* __restrict__ Wd,  // [H,1]
    const float* __restrict__ bd,  // [1]
    float* __restrict__ out)       // [B,1]
{
    __shared__ _Float16 hbuf[2][16 * HQ];   // h tile, double buffered (8 KB)
    __shared__ float    xbuf[16 * CHUNK];   // x chunk (8 KB)

    const int tid  = threadIdx.x;
    const int w    = tid >> 5;        // wave id 0..7
    const int lane = tid & 31;
    const int hi   = lane >> 4;       // half-wave
    const int ln   = lane & 15;
    const int j    = w * 16 + ln;     // hidden column owned by this lane
    const int wg16 = blockIdx.x * 16; // first batch row of this WG

    // ---- init: zero h(t=0) buffer ----
    for (int i = tid; i < 16 * HQ; i += 256) hbuf[0][i] = (_Float16)0.0f;

    // ---- load Wh as WMMA B fragments (f32 -> f16), held in regs forever ----
    // B fragment k order used here, same as A (32x16): lanes 0-15: K=0..15, lanes 16-31: K=16..31,
    // column = lane&15, K ascending across the 16 halves.
    v16h bfrag[4][4];
#pragma unroll
    for (int g = 0; g < 4; ++g) {
        const int n = g * HQ + j;
#pragma unroll
        for (int kc = 0; kc < 4; ++kc) {
            const int kb = kc * 32 + hi * 16;
#pragma unroll
            for (int m = 0; m < 16; ++m)
                bfrag[g][kc][m] = (_Float16)Wh[(kb + m) * G4H + n];
        }
    }

    // per-lane input-gate constants for owned columns (gate order i,f,g,o)
    float wi_r[4], b_r[4];
#pragma unroll
    for (int g = 0; g < 4; ++g) {
        wi_r[g] = Wi[g * HQ + j];
        b_r[g]  = b[g * HQ + j];
    }

    // cell state in registers (C-fragment layout: row M = r + hi*8, col j)
    float c_r[8];
#pragma unroll
    for (int r = 0; r < 8; ++r) c_r[r] = 0.0f;

    int p = 0;
    for (int t = 0; t < SQ; ++t) {
        const int tc = t & (CHUNK - 1);
        if (tc == 0) {
            // refill x chunk: 16 rows x 128 steps, coalesced along t
            for (int i = tid; i < 16 * CHUNK; i += 256) {
                const int rr = i >> 7;           // row in tile
                const int tt = i & (CHUNK - 1);  // step in chunk
                xbuf[i] = x[(wg16 + rr) * SQ + t + tt];
            }
            __syncthreads();
        }

        // ---- A fragments (h tile) from LDS, A 16x32 f16 layout ----
        // lane row = ln; halves: {0..15} (lo half-wave) / {16..31}
        union { v16h v; v8h h8[2]; } afr[4];
#pragma unroll
        for (int kc = 0; kc < 4; ++kc) {
            const int base = ln * HQ + kc * 32 + hi * 16;
            afr[kc].h8[0] = *(const v8h*)&hbuf[p][base];
            afr[kc].h8[1] = *(const v8h*)&hbuf[p][base + 8];
        }

        // ---- gates = h @ Wh : 16 WMMAs, f32 accumulate ----
        v8f acc[4] = { {}, {}, {}, {} };
#pragma unroll
        for (int kc = 0; kc < 4; ++kc) {
#pragma unroll
            for (int g = 0; g < 4; ++g) {
                acc[g] = __builtin_amdgcn_wmma_f32_16x16x32_f16(
                    false, afr[kc].v, false, bfrag[g][kc],
                    (short)0, acc[g], false, false);
            }
        }

        // ---- elementwise LSTM cell update (f32) ----
#pragma unroll
        for (int r = 0; r < 8; ++r) {
            const int M = r + hi * 8;                  // batch row in tile
            const float xv = xbuf[M * CHUNK + tc];
            const float gi = acc[0][r] + xv * wi_r[0] + b_r[0];
            const float gf = acc[1][r] + xv * wi_r[1] + b_r[1];
            const float gg = acc[2][r] + xv * wi_r[2] + b_r[2];
            const float go = acc[3][r] + xv * wi_r[3] + b_r[3];
            const float iv = fast_sigmoid(gi);
            const float fv = fast_sigmoid(gf);
            const float gv = fast_tanh(gg);
            const float ov = fast_sigmoid(go);
            const float cn = fv * c_r[r] + iv * gv;
            c_r[r] = cn;
            const float hn = ov * fast_tanh(cn);
            hbuf[p ^ 1][M * HQ + j] = (_Float16)hn;
        }

        __syncthreads();
        p ^= 1;
    }

    // ---- final projection: out[row] = h_f . Wd + bd ----
    if (tid < 16) {
        float sum = 0.0f;
        for (int k = 0; k < HQ; ++k)
            sum += (float)hbuf[p][tid * HQ + k] * Wd[k];
        out[wg16 + tid] = sum + bd[0];
    }
}

extern "C" void kernel_launch(void* const* d_in, const int* in_sizes, int n_in,
                              void* d_out, int out_size, void* d_ws, size_t ws_size,
                              hipStream_t stream) {
    const float* x  = (const float*)d_in[0];
    const float* Wi = (const float*)d_in[1];
    const float* Wh = (const float*)d_in[2];
    const float* b  = (const float*)d_in[3];
    const float* Wd = (const float*)d_in[4];
    const float* bd = (const float*)d_in[5];
    float* out = (float*)d_out;

    lstm_wmma_kernel<<<BQ / 16, 256, 0, stream>>>(x, Wi, Wh, b, Wd, bd, out);
}
